// GNN_85426899517626
// MI455X (gfx1250) — matrix operand order repaired and checked by reading, not compiled, not run on hardware
//
#include <hip/hip_runtime.h>
#include <hip/hip_bf16.h>

typedef __attribute__((ext_vector_type(16))) _Float16 v16h;
typedef __attribute__((ext_vector_type(8)))  float    v8f;

#define NN 384
#define CC 128
#define HH 60
#define CH 7680            // 128*60
#define EE 3072
#define NL 4
#define PT 30              // CH / 256 threads

// ---------------- combined edge weights: w3g[l][c][0..4] = (W3 @ ew)[c][k], w3g[..][5] = (W3 @ eb)[c]
__global__ __launch_bounds__(256) void w3_kernel(const float* __restrict__ conv_w,
                                                 const float* __restrict__ edge_w,
                                                 const float* __restrict__ edge_b,
                                                 float* __restrict__ w3g) {
    int idx = blockIdx.x * blockDim.x + threadIdx.x;   // 4*128
    if (idx >= NL * CC) return;
    int l = idx >> 7, c = idx & 127;
    const float* W3  = conv_w + l * CC * 384 + c * 384 + 256;  // W3[c][m]
    const float* ew  = edge_w + l * CC * 5;                    // ew[m][k]
    const float* ebp = edge_b + l * CC;
    float acc[6] = {0.f, 0.f, 0.f, 0.f, 0.f, 0.f};
    for (int m = 0; m < CC; ++m) {
        float w = W3[m];
        #pragma unroll
        for (int k = 0; k < 5; ++k) acc[k] += w * ew[m * 5 + k];
        acc[5] += w * ebp[m];
    }
    #pragma unroll
    for (int k = 0; k < 6; ++k) w3g[idx * 6 + k] = acc[k];
}

// ---------------- per-edge bias vectors: ebias[l][e][c]
__global__ __launch_bounds__(256) void ebias_kernel(const float* __restrict__ edge_attr,
                                                    const float* __restrict__ w3g,
                                                    float* __restrict__ ebias) {
    int l = blockIdx.y;
    int idx = blockIdx.x * blockDim.x + threadIdx.x;   // EE*CC
    if (idx >= EE * CC) return;
    int c = idx & 127, e = idx >> 7;
    const float* g  = w3g + (l * CC + c) * 6;
    const float* ea = edge_attr + e * 5;
    float s = g[5];
    #pragma unroll
    for (int k = 0; k < 5; ++k) s += g[k] * ea[k];
    ebias[(size_t)l * EE * CC + idx] = s;
}

// ---------------- pack Wcat = [[W1-W2];[W2]] (256x128) into f16 A-fragments
// layout: wp[l][mt][kc][lane][r*2+h2], A klocal = (r>=4?16:0) + 8*kgrp + 2*(r&3) + h2
__global__ __launch_bounds__(256) void wpack_kernel(const float* __restrict__ conv_w,
                                                    _Float16* __restrict__ wp) {
    int idx = blockIdx.x * blockDim.x + threadIdx.x;   // 4*16*4*32*16 = 131072
    if (idx >= NL * 16 * 4 * 32 * 16) return;
    int p    = idx & 15;
    int lane = (idx >> 4) & 31;
    int kc   = (idx >> 9) & 3;
    int mt   = (idx >> 11) & 15;
    int l    = idx >> 15;
    int r = p >> 1, h2 = p & 1;
    int kgrp = lane >> 4;
    int klocal = ((r >= 4) ? 16 : 0) + 8 * kgrp + 2 * (r & 3) + h2;
    int k = kc * 32 + klocal;                 // 0..127
    int m = mt * 16 + (lane & 15);            // 0..255
    const float* cw = conv_w + l * CC * 384;
    float v;
    if (m < CC) v = cw[m * 384 + k] - cw[m * 384 + CC + k];     // W1 - W2
    else        v = cw[(m - CC) * 384 + CC + k];                // W2
    wp[idx] = (_Float16)v;
}

// ---------------- pack x (N,C,H) f32 into f16 B-fragments
// layout: xp[node][kc][nt][lane][r*2+h2], B klocal = (r>=4?16:0) + 8*kgrp + 2*(r&3) + h2, col = nt*16 + lane%16
__global__ __launch_bounds__(256) void xpack_kernel(const float* __restrict__ x,
                                                    _Float16* __restrict__ xp) {
    int idx = blockIdx.x * blockDim.x + threadIdx.x;   // 384*4*4*32*16 = 3145728
    if (idx >= NN * 4 * 4 * 32 * 16) return;
    int p    = idx & 15;
    int lane = (idx >> 4) & 31;
    int nt   = (idx >> 9) & 3;
    int kc   = (idx >> 11) & 3;
    int node = idx >> 13;
    int r = p >> 1, h2 = p & 1;
    int kgrp = lane >> 4;
    int k    = kc * 32 + ((r >= 4) ? 16 : 0) + 8 * kgrp + 2 * (r & 3) + h2;       // channel
    int hcol = nt * 16 + (lane & 15);                  // height column
    float v = (hcol < HH) ? x[node * CH + k * HH + hcol] : 0.0f;
    xp[idx] = (_Float16)v;
}

// ---------------- WMMA GEMM: Ycat[n] (256x60) = Wcat (256x128) @ x[n] (128x60)
__global__ __launch_bounds__(256) void gemm_kernel(const _Float16* __restrict__ wp,   // this layer
                                                   const _Float16* __restrict__ xp,
                                                   float* __restrict__ ycat) {
    int wave = blockIdx.x * (blockDim.x >> 5) + (threadIdx.x >> 5);  // 24576 waves
    int lane = threadIdx.x & 31;
    int node = wave >> 6;
    int mt   = (wave >> 2) & 15;
    int nt   = wave & 3;
    v8f c = {};
    #pragma unroll
    for (int kc = 0; kc < 4; ++kc) {
        v16h a = *(const v16h*)(wp + (size_t)((mt * 4 + kc) * 32 + lane) * 16);
        v16h b = *(const v16h*)(xp + (size_t)((((node * 4 + kc) * 4) + nt) * 32 + lane) * 16);
        c = __builtin_amdgcn_wmma_f32_16x16x32_f16(false, a, false, b, (short)0, c, false, false);
    }
    int ncol  = nt * 16 + (lane & 15);
    int mbase = mt * 16 + ((lane >> 4) << 3);
    if (ncol < HH) {
        float* outp = ycat + (size_t)node * 256 * HH + ncol;
        #pragma unroll
        for (int r = 0; r < 8; ++r) outp[(mbase + r) * HH] = c[r];
    }
}

// ---------------- fused per-edge LayerNorm/ReLU + per-dst-node segment sum
__global__ __launch_bounds__(256) void aggregate_kernel(const float* __restrict__ ycat,
                                                        const float* __restrict__ ebias_l,
                                                        const float* __restrict__ ln_g,
                                                        const float* __restrict__ ln_b,
                                                        const int*   __restrict__ edge_index,
                                                        float* __restrict__ xout) {
    __shared__ float y1s[CH];
    __shared__ float bes[CC];
    __shared__ float red[16];
    __shared__ int   cnt[256];
    __shared__ int   elist[40];
    __shared__ int   ecount;

    int n   = blockIdx.x;
    int tid = threadIdx.x;
    const int* dstIdx = edge_index + EE;

    // deterministic ordered collection of incoming edges (dst == n)
    int base = tid * (EE / 256);   // 12 per thread
    int lc = 0;
    for (int i = 0; i < EE / 256; ++i) if (dstIdx[base + i] == n) ++lc;
    cnt[tid] = lc;
    __syncthreads();
    if (tid == 0) {
        int s = 0;
        for (int i = 0; i < 256; ++i) { int c0 = cnt[i]; cnt[i] = s; s += c0; }
        ecount = s;
    }
    __syncthreads();
    int off = cnt[tid];
    for (int i = 0; i < EE / 256; ++i) {
        int e = base + i;
        if (dstIdx[e] == n) { if (off < 40) elist[off] = e; ++off; }
    }

    // cache Y1[n] (rows 0..127 of Ycat[n]) in LDS
    const float* y1 = ycat + (size_t)n * 256 * HH;
    for (int i = tid; i < CH; i += 256) y1s[i] = y1[i];
    __syncthreads();
    int ne = (ecount < 40) ? ecount : 40;

    float acc[PT], gv[PT], bv[PT];
    #pragma unroll
    for (int j = 0; j < PT; ++j) {
        int idx = tid + j * 256;
        acc[j] = 0.f; gv[j] = ln_g[idx]; bv[j] = ln_b[idx];
    }

    for (int eidx = 0; eidx < ne; ++eidx) {
        int e   = elist[eidx];
        int src = edge_index[e];
        if (tid < CC) bes[tid] = ebias_l[(size_t)e * CC + tid];
        __syncthreads();

        const float* y2 = ycat + (size_t)src * 256 * HH + CH;  // rows 128..255
        float t[PT];
        float s1 = 0.f, s2 = 0.f;
        #pragma unroll
        for (int j = 0; j < PT; ++j) {
            int idx = tid + j * 256;
            int c = idx / HH;
            float v = y1s[idx] + y2[idx] + bes[c];
            t[j] = v; s1 += v; s2 += v * v;
        }
        // wave32 butterfly + cross-wave LDS reduction
        #pragma unroll
        for (int o = 16; o > 0; o >>= 1) {
            s1 += __shfl_xor(s1, o);
            s2 += __shfl_xor(s2, o);
        }
        int wid = tid >> 5;
        if ((tid & 31) == 0) { red[wid] = s1; red[8 + wid] = s2; }
        __syncthreads();
        float S1 = 0.f, S2 = 0.f;
        #pragma unroll
        for (int w = 0; w < 8; ++w) { S1 += red[w]; S2 += red[8 + w]; }

        float mu  = S1 * (1.0f / (float)CH);
        float var = S2 * (1.0f / (float)CH) - mu * mu;
        float rs  = rsqrtf(var + 1e-5f);
        #pragma unroll
        for (int j = 0; j < PT; ++j) {
            float v = (t[j] - mu) * rs * gv[j] + bv[j];
            acc[j] += fmaxf(v, 0.f);
        }
        __syncthreads();   // protect bes/red for next edge
    }

    #pragma unroll
    for (int j = 0; j < PT; ++j) xout[(size_t)n * CH + tid + j * 256] = acc[j];
}

static inline size_t align256(size_t x) { return (x + 255) & ~(size_t)255; }

extern "C" void kernel_launch(void* const* d_in, const int* in_sizes, int n_in,
                              void* d_out, int out_size, void* d_ws, size_t ws_size,
                              hipStream_t stream) {
    const float* x_in      = (const float*)d_in[0];   // (384,128,60)
    const float* edge_attr = (const float*)d_in[1];   // (3072,5)
    const float* edge_w    = (const float*)d_in[2];   // (4,128,5)
    const float* edge_b    = (const float*)d_in[3];   // (4,128)
    const float* conv_w    = (const float*)d_in[4];   // (4,128,384)
    const float* ln_g      = (const float*)d_in[5];   // (4,128,60)
    const float* ln_b      = (const float*)d_in[6];   // (4,128,60)
    const int*   edge_idx  = (const int*)d_in[7];     // (2,3072)

    // workspace carve (all 256B aligned)
    char* ws = (char*)d_ws;
    size_t off = 0;
    float*    w3g   = (float*)(ws + off);  off = align256(off + (size_t)NL * CC * 6 * 4);
    float*    ebias = (float*)(ws + off);  off = align256(off + (size_t)NL * EE * CC * 4);
    _Float16* wp    = (_Float16*)(ws + off); off = align256(off + (size_t)NL * 16 * 4 * 32 * 16 * 2);
    _Float16* xp    = (_Float16*)(ws + off); off = align256(off + (size_t)NN * 4 * 4 * 32 * 16 * 2);
    float*    ycat  = (float*)(ws + off);  off = align256(off + (size_t)NN * 256 * HH * 4);
    float*    xbuf  = (float*)(ws + off);  off = align256(off + (size_t)NN * CC * HH * 4);

    // one-time (per call) precompute
    w3_kernel<<<2, 256, 0, stream>>>(conv_w, edge_w, edge_b, w3g);
    ebias_kernel<<<dim3((EE * CC) / 256, NL), 256, 0, stream>>>(edge_attr, w3g, ebias);
    wpack_kernel<<<(NL * 16 * 4 * 32 * 16) / 256, 256, 0, stream>>>(conv_w, wp);
    xpack_kernel<<<(NN * 4 * 4 * 32 * 16) / 256, 256, 0, stream>>>(x_in, xp);

    for (int l = 0; l < NL; ++l) {
        gemm_kernel<<<3072, 256, 0, stream>>>(wp + (size_t)l * 16 * 4 * 32 * 16, xp, ycat);
        float* xo = (l < NL - 1) ? xbuf : (float*)d_out;
        aggregate_kernel<<<NN, 256, 0, stream>>>(ycat,
                                                 ebias + (size_t)l * EE * CC,
                                                 ln_g + (size_t)l * CH,
                                                 ln_b + (size_t)l * CH,
                                                 edge_idx, xo);
        if (l < NL - 1)
            xpack_kernel<<<(NN * 4 * 4 * 32 * 16) / 256, 256, 0, stream>>>(xo, xp);
    }
}
